// LanguageModel_19782619365686
// MI455X (gfx1250) — matrix operand order repaired and checked by reading, not compiled, not run on hardware
//
#include <hip/hip_runtime.h>
#include <hip/hip_bf16.h>
#include <math.h>

// ---------------- problem constants ----------------
#define BB 32
#define TT 60
#define FF 4096
#define PP 512
#define HH 512
#define VV 10000
#define NW 30
#define START_ID 1
#define BT (BB*TT)          // 1920
#define KCAT (PP+HH)        // 1024
#define NGATE (4*HH)        // 2048

// ---------------- types for WMMA ----------------
typedef __bf16 bfx16 __attribute__((ext_vector_type(16)));
typedef float  v8f   __attribute__((ext_vector_type(8)));

union FragU { uint4 u[2]; bfx16 v; };

__device__ __forceinline__ unsigned short f32_to_bf16(float f) {
    unsigned int u = __float_as_uint(f);
    u += 0x7FFFu + ((u >> 16) & 1u);          // round-to-nearest-even
    return (unsigned short)(u >> 16);
}

__device__ __forceinline__ float sigmoidf_(float x) { return 1.0f / (1.0f + expf(-x)); }

// A fragment: 16x32 bf16, row-major source [M,K].
// lane 0-15 : row = lane,     K = k0+{0..7} and k0+{16..23}
// lane 16-31: row = lane-16,  K = k0+{8..15} and k0+{24..31}
__device__ __forceinline__ bfx16 load_a_frag(const unsigned short* A, int ldk,
                                             int row0, int lane, int k0) {
    int m = lane & 15;
    int koff = k0 + ((lane >> 4) << 3);
    const uint4* p = reinterpret_cast<const uint4*>(A + (size_t)(row0 + m) * ldk + koff);
    FragU f; f.u[0] = p[0]; f.u[1] = p[2];    // +16 elements = +2 uint4
    return f.v;
}

// B fragment: 32x16 bf16, source is W[N,K] row-major (i.e. B transposed).
// lane 0-15 : col = lane,     K = k0+{0..7} and k0+{16..23}
// lane 16-31: col = lane-16,  K = k0+{8..15} and k0+{24..31}
__device__ __forceinline__ bfx16 load_b_frag(const unsigned short* W, int ldk,
                                             int col0, int lane, int k0) {
    int n = lane & 15;
    int koff = k0 + ((lane >> 4) << 3);
    const uint4* p = reinterpret_cast<const uint4*>(W + (size_t)(col0 + n) * ldk + koff);
    FragU f; f.u[0] = p[0]; f.u[1] = p[2];
    return f.v;
}

// C[m,n] = act( sum_k A[m,k]*W[n,k] + bias[n] ), A:[M,K] bf16, W:[N,K] bf16.
// One wave -> 32x16 slab of C (two accumulators share one B fragment).
// M%32==0, N%16==0, K%32==0 (true for every GEMM in this model).
__global__ void gemm_bf16_wmma(const unsigned short* __restrict__ A,
                               const unsigned short* __restrict__ W,
                               const float* __restrict__ bias,
                               float* __restrict__ C,
                               unsigned short* __restrict__ Cbf,
                               int M, int N, int K, int ldc, int act) {
    int lane = threadIdx.x & 31;
    int wave = threadIdx.x >> 5;
    int tile = blockIdx.x * (blockDim.x >> 5) + wave;
    int nt = N >> 4;
    int tot = (M >> 5) * nt;
    if (tile >= tot) return;
    int row0 = (tile / nt) << 5;
    int col0 = (tile % nt) << 4;

    v8f acc0 = {0.f,0.f,0.f,0.f,0.f,0.f,0.f,0.f};
    v8f acc1 = {0.f,0.f,0.f,0.f,0.f,0.f,0.f,0.f};

    for (int k0 = 0; k0 < K; k0 += 32) {
        bfx16 a0 = load_a_frag(A, K, row0,      lane, k0);
        bfx16 a1 = load_a_frag(A, K, row0 + 16, lane, k0);
        bfx16 b  = load_b_frag(W, K, col0,      lane, k0);
        acc0 = __builtin_amdgcn_wmma_f32_16x16x32_bf16(false, a0, false, b,
                                                       (short)0, acc0, false, false);
        acc1 = __builtin_amdgcn_wmma_f32_16x16x32_bf16(false, a1, false, b,
                                                       (short)0, acc1, false, false);
    }

    int n  = col0 + (lane & 15);
    int mb = row0 + ((lane >> 4) << 3);       // lanes>=16 handle rows +8
    float bv = bias ? bias[n] : 0.0f;
#pragma unroll
    for (int i = 0; i < 8; ++i) {
        float x0 = acc0[i] + bv;
        float x1 = acc1[i] + bv;
        if (act) { x0 = tanhf(x0); x1 = tanhf(x1); }
        C[(size_t)(mb + i) * ldc + n]      = x0;
        C[(size_t)(mb + 16 + i) * ldc + n] = x1;
        if (Cbf) {
            Cbf[(size_t)(mb + i) * ldc + n]      = f32_to_bf16(x0);
            Cbf[(size_t)(mb + 16 + i) * ldc + n] = f32_to_bf16(x1);
        }
    }
}

// ---------------- small helper kernels ----------------
__global__ void cvt_f32_bf16(const float* __restrict__ s, unsigned short* __restrict__ d, int n) {
    int i = blockIdx.x * blockDim.x + threadIdx.x;
    if (i < n) d[i] = f32_to_bf16(s[i]);
}

// src [R,C] fp32 -> dst [C,R] bf16
__global__ void transpose_f32_bf16(const float* __restrict__ s, unsigned short* __restrict__ d,
                                   int R, int C) {
    int i = blockIdx.x * blockDim.x + threadIdx.x;
    if (i < R * C) {
        int r = i / C, c = i % C;
        d[(size_t)c * R + r] = f32_to_bf16(s[i]);
    }
}

// Wcat[j, 0:512]=W_ih[j,:], Wcat[j,512:1024]=W_hh[j,:]  (bf16)
__global__ void build_wcat(const float* __restrict__ Wih, const float* __restrict__ Whh,
                           unsigned short* __restrict__ Wcat) {
    int i = blockIdx.x * blockDim.x + threadIdx.x;
    if (i < NGATE * KCAT) {
        int j = i / KCAT, k = i % KCAT;
        float v = (k < PP) ? Wih[j * PP + k] : Whh[j * HH + (k - PP)];
        Wcat[i] = f32_to_bf16(v);
    }
}

__global__ void build_bias(const float* __restrict__ bih, const float* __restrict__ bhh,
                           float* __restrict__ bcat) {
    int i = blockIdx.x * blockDim.x + threadIdx.x;
    if (i < NGATE) bcat[i] = bih[i] + bhh[i];
}

// mean over T frames: mean[b,p] = (1/60) sum_t v[b,t,p]
__global__ void mean_kernel(const float* __restrict__ v, float* __restrict__ mean) {
    int b = blockIdx.x, p = threadIdx.x;     // 512 threads
    float s = 0.f;
    for (int t = 0; t < TT; ++t) s += v[((size_t)b * TT + t) * PP + p];
    mean[b * PP + p] = s * (1.0f / TT);
}

// out = tanh(tanh(tanh(x@W1+b1)@W2+b2))   W1,W2:[512,512] col-indexed [k,n]
__global__ void init_mlp(const float* __restrict__ x,
                         const float* __restrict__ W1, const float* __restrict__ b1,
                         const float* __restrict__ W2, const float* __restrict__ b2,
                         float* __restrict__ out) {
    __shared__ float sx[HH], sy[HH];
    int b = blockIdx.x, n = threadIdx.x;     // 512 threads
    sx[n] = x[b * PP + n];
    __syncthreads();
    float s = b1[n];
    for (int k = 0; k < PP; ++k) s += sx[k] * W1[k * HH + n];
    sy[n] = tanhf(s);
    __syncthreads();
    float s2 = b2[n];
    for (int k = 0; k < HH; ++k) s2 += sy[k] * W2[k * HH + n];
    out[b * HH + n] = tanhf(tanhf(s2));
}

__global__ void init_word(const float* __restrict__ embed, float* __restrict__ word) {
    int i = blockIdx.x * blockDim.x + threadIdx.x;
    if (i < BB * PP) word[i] = embed[(size_t)START_ID * PP + (i % PP)];
}

// Fused per-step attention: hWh, scores, softmax, context; emits Acat=[t|h] bf16
__global__ void attention_kernel(const float* __restrict__ h, const float* __restrict__ vs,
                                 const float* __restrict__ v, const float* __restrict__ Wh,
                                 const float* __restrict__ aw, const float* __restrict__ word,
                                 unsigned short* __restrict__ Acat) {
    __shared__ float sh_h[HH], sh_w[HH], sh_e[64];
    int b = blockIdx.x, tid = threadIdx.x;   // 256 threads
    for (int p = tid; p < HH; p += 256) sh_h[p] = h[b * HH + p];
    __syncthreads();
    for (int n = tid; n < HH; n += 256) {
        float s = 0.f;
        for (int k = 0; k < HH; ++k) s += sh_h[k] * Wh[k * HH + n];
        sh_w[n] = s;
    }
    __syncthreads();
    if (tid < TT) {
        const float* vr = vs + ((size_t)b * TT + tid) * HH;
        float s = 0.f;
        for (int k = 0; k < HH; ++k) s += tanhf(vr[k] + sh_w[k]) * aw[k];
        sh_e[tid] = s;
    }
    __syncthreads();
    if (tid == 0) {
        float m = -3.402823466e38f;
        for (int t = 0; t < TT; ++t) m = fmaxf(m, sh_e[t]);
        float sum = 0.f;
        for (int t = 0; t < TT; ++t) { float ex = expf(sh_e[t] - m); sh_e[t] = ex; sum += ex; }
        float inv = 1.0f / sum;
        for (int t = 0; t < TT; ++t) sh_e[t] *= inv;
    }
    __syncthreads();
    for (int p = tid; p < PP; p += 256) {
        float ctx = 0.f;
        for (int t = 0; t < TT; ++t) ctx += sh_e[t] * v[((size_t)b * TT + t) * PP + p];
        float tv = word[b * PP + p] + ctx;
        Acat[(size_t)b * KCAT + p]      = f32_to_bf16(tv);
        Acat[(size_t)b * KCAT + PP + p] = f32_to_bf16(sh_h[p]);
    }
}

// gates [B,2048] (i,f,g,o) -> update h,c; also emit h bf16 for logits GEMM
__global__ void lstm_update(const float* __restrict__ gates, float* __restrict__ h,
                            float* __restrict__ c, unsigned short* __restrict__ hbf) {
    int b = blockIdx.x, j = threadIdx.x;     // 512 threads
    const float* g = gates + (size_t)b * NGATE;
    float ig = g[j], fg = g[HH + j], gg = g[2 * HH + j], og = g[3 * HH + j];
    float cp = c[b * HH + j];
    float c2 = sigmoidf_(fg) * cp + sigmoidf_(ig) * tanhf(gg);
    float h2 = sigmoidf_(og) * tanhf(c2);
    c[b * HH + j] = c2;
    h[b * HH + j] = h2;
    hbf[b * HH + j] = f32_to_bf16(h2);
}

// first-occurrence argmax over V, then word = embed[wid]
__global__ void argmax_embed(const float* __restrict__ logitsBase, long rowStride,
                             const float* __restrict__ embed, float* __restrict__ word) {
    __shared__ float sv[256];
    __shared__ int   si[256];
    int b = blockIdx.x, tid = threadIdx.x;   // 256 threads
    const float* row = logitsBase + (size_t)b * rowStride;
    float best = -3.402823466e38f; int bi = 0x7fffffff;
    for (int j = tid; j < VV; j += 256) {
        float x = row[j];
        if (x > best || (x == best && j < bi)) { best = x; bi = j; }
    }
    sv[tid] = best; si[tid] = bi;
    __syncthreads();
    for (int s = 128; s > 0; s >>= 1) {
        if (tid < s) {
            float v2 = sv[tid + s]; int i2 = si[tid + s];
            if (v2 > sv[tid] || (v2 == sv[tid] && i2 < si[tid])) { sv[tid] = v2; si[tid] = i2; }
        }
        __syncthreads();
    }
    int wid = si[0];
    for (int p = tid; p < PP; p += 256) word[b * PP + p] = embed[(size_t)wid * PP + p];
}

// ---------------- host orchestration ----------------
extern "C" void kernel_launch(void* const* d_in, const int* in_sizes, int n_in,
                              void* d_out, int out_size, void* d_ws, size_t ws_size,
                              hipStream_t stream) {
    const float* video  = (const float*)d_in[0];
    // d_in[1] captions: unused (ss_prob==0 -> pure argmax feedback)
    const float* enc_W  = (const float*)d_in[2];
    const float* enc_b  = (const float*)d_in[3];
    const float* att_Wh = (const float*)d_in[4];
    const float* att_Uv = (const float*)d_in[5];
    const float* att_w  = (const float*)d_in[6];
    const float* embed  = (const float*)d_in[7];
    const float* W_ih   = (const float*)d_in[8];
    const float* W_hh   = (const float*)d_in[9];
    const float* b_ih   = (const float*)d_in[10];
    const float* b_hh   = (const float*)d_in[11];
    const float* ih_W1  = (const float*)d_in[12];
    const float* ih_b1  = (const float*)d_in[13];
    const float* ih_W2  = (const float*)d_in[14];
    const float* ih_b2  = (const float*)d_in[15];
    const float* ic_W1  = (const float*)d_in[16];
    const float* ic_b1  = (const float*)d_in[17];
    const float* ic_W2  = (const float*)d_in[18];
    const float* ic_b2  = (const float*)d_in[19];
    const float* out_W  = (const float*)d_in[20];
    const float* out_b  = (const float*)d_in[21];
    float* out = (float*)d_out;

    // workspace carve (256B aligned)
    char* ws = (char*)d_ws;
    size_t off = 0;
    auto carve = [&](size_t bytes) -> void* {
        void* p = ws + off;
        off = (off + bytes + 255) & ~(size_t)255;
        return p;
    };
    unsigned short* vid_bf   = (unsigned short*)carve((size_t)BT * FF * 2);
    unsigned short* encWt_bf = (unsigned short*)carve((size_t)PP * FF * 2);
    float*          v_f32    = (float*)carve((size_t)BT * PP * 4);
    unsigned short* v_bf     = (unsigned short*)carve((size_t)BT * PP * 2);
    unsigned short* uvT_bf   = (unsigned short*)carve((size_t)HH * PP * 2);
    float*          vs_f32   = (float*)carve((size_t)BT * HH * 4);
    unsigned short* outW_bf  = (unsigned short*)carve((size_t)VV * HH * 2);
    unsigned short* Wcat_bf  = (unsigned short*)carve((size_t)NGATE * KCAT * 2);
    float*          bcat     = (float*)carve((size_t)NGATE * 4);
    float*          mean_v   = (float*)carve((size_t)BB * PP * 4);
    float*          h_st     = (float*)carve((size_t)BB * HH * 4);
    float*          c_st     = (float*)carve((size_t)BB * HH * 4);
    unsigned short* h_bf     = (unsigned short*)carve((size_t)BB * HH * 2);
    unsigned short* Acat     = (unsigned short*)carve((size_t)BB * KCAT * 2);
    float*          gates    = (float*)carve((size_t)BB * NGATE * 4);
    float*          word     = (float*)carve((size_t)BB * PP * 4);

    auto cdiv = [](int a, int b) { return (a + b - 1) / b; };

    // ---- one-time operand prep (bf16 + weight transposes) ----
    cvt_f32_bf16<<<cdiv(BT * FF, 256), 256, 0, stream>>>(video, vid_bf, BT * FF);
    transpose_f32_bf16<<<cdiv(FF * PP, 256), 256, 0, stream>>>(enc_W, encWt_bf, FF, PP);
    transpose_f32_bf16<<<cdiv(PP * HH, 256), 256, 0, stream>>>(att_Uv, uvT_bf, PP, HH);
    cvt_f32_bf16<<<cdiv(VV * HH, 256), 256, 0, stream>>>(out_W, outW_bf, VV * HH);
    build_wcat<<<cdiv(NGATE * KCAT, 256), 256, 0, stream>>>(W_ih, W_hh, Wcat_bf);
    build_bias<<<cdiv(NGATE, 256), 256, 0, stream>>>(b_ih, b_hh, bcat);

    // ---- encoder: v = tanh(X @ enc_W + b)  [1920,4096]x[4096,512] ----
    {
        int tot = (BT >> 5) * (PP >> 4);
        gemm_bf16_wmma<<<cdiv(tot, 4), 128, 0, stream>>>(
            vid_bf, encWt_bf, enc_b, v_f32, v_bf, BT, PP, FF, PP, /*act=tanh*/1);
    }
    // ---- vs_ = v @ att_Uv  [1920,512]x[512,512] ----
    {
        int tot = (BT >> 5) * (HH >> 4);
        gemm_bf16_wmma<<<cdiv(tot, 4), 128, 0, stream>>>(
            v_bf, uvT_bf, nullptr, vs_f32, nullptr, BT, HH, PP, HH, 0);
    }
    // ---- init state ----
    mean_kernel<<<BB, PP, 0, stream>>>(v_f32, mean_v);
    init_mlp<<<BB, HH, 0, stream>>>(mean_v, ih_W1, ih_b1, ih_W2, ih_b2, h_st);
    init_mlp<<<BB, HH, 0, stream>>>(mean_v, ic_W1, ic_b1, ic_W2, ic_b2, c_st);
    init_word<<<cdiv(BB * PP, 256), 256, 0, stream>>>(embed, word);

    // ---- 30 sequential decode steps ----
    const long ldcOut = (long)NW * VV;
    for (int s = 0; s < NW; ++s) {
        attention_kernel<<<BB, 256, 0, stream>>>(h_st, vs_f32, v_f32, att_Wh, att_w,
                                                 word, Acat);
        {   // gates = [t|h] @ [W_ih|W_hh]^T + (b_ih+b_hh)   [32,1024]x[1024,2048]
            int tot = (BB >> 5) * (NGATE >> 4);
            gemm_bf16_wmma<<<cdiv(tot, 4), 128, 0, stream>>>(
                Acat, Wcat_bf, bcat, gates, nullptr, BB, NGATE, KCAT, NGATE, 0);
        }
        lstm_update<<<BB, HH, 0, stream>>>(gates, h_st, c_st, h_bf);
        {   // logits = h2 @ out_W^T + out_b   [32,512]x[512,10000] -> d_out[b,s,:]
            int tot = (BB >> 5) * (VV >> 4);
            gemm_bf16_wmma<<<cdiv(tot, 4), 128, 0, stream>>>(
                h_bf, outW_bf, out_b, out + (size_t)s * VV, nullptr,
                BB, VV, HH, (int)ldcOut, 0);
        }
        argmax_embed<<<BB, 256, 0, stream>>>(out + (size_t)s * VV, ldcOut, embed, word);
    }
}
